// GCNNet_88459146428507
// MI455X (gfx1250) — matrix operand order repaired and checked by reading, not compiled, not run on hardware
//
#include <hip/hip_runtime.h>
#include <hip/hip_bf16.h>

// ---------------------------------------------------------------------------
// GCNNet forward for MI455X (gfx1250, wave32, WMMA).
//
// Roofline: heavy parts are the bidirectional LSTM recurrence (33.6 GFLOP,
// sequential over T=1000, batch-parallel) and fc_xt whose 131MB weight is the
// dominant byte stream (~5.6us at 23.3 TB/s HBM). xseq@Wih.T collapses to a
// 26x512 embedding-projection table because xseq = emb[target].
// LSTM kernel: Whh resident in LDS (f16) for the whole sequence; gates via
// v_wmma_f32_16x16x32_f16 (M=16 batch chunk, N=512, K=128); fc_xt fused per
// step (lstm_out never materialized; fc_xt_w streamed from HBM exactly once,
// L2-shared across the 8 batch-chunk WGs per direction, prefetched one step
// ahead so the serial recurrence never stalls on it).
// ---------------------------------------------------------------------------

typedef __attribute__((ext_vector_type(16))) _Float16 v16h;
typedef __attribute__((ext_vector_type(8)))  float    v8f;

#define N_NODES 5120
#define NPG     40      // nodes per graph
#define BATCH   128
#define TSEQ    1000
#define HID     128
#define GDIM    512     // 4*H

// ----------------------------- small kernels -------------------------------

__global__ void k_fill(float* p, float v, int n) {
    int i = blockIdx.x * blockDim.x + threadIdx.x;
    if (i < n) p[i] = v;
}

__global__ void k_deg(const int* dst, int nE, float* deg) {
    int i = blockIdx.x * blockDim.x + threadIdx.x;
    if (i < nE) atomicAdd(&deg[dst[i]], 1.0f);
}

__global__ void k_rsqrt(float* p, int n) {
    int i = blockIdx.x * blockDim.x + threadIdx.x;
    if (i < n) p[i] = rsqrtf(fmaxf(p[i], 1.0e-12f));
}

// C[m,n] = act( sum_k A[m,k]*B[k,n] + bias[n] )
__global__ void k_gemm(const float* __restrict__ A, const float* __restrict__ B,
                       const float* __restrict__ bias, float* __restrict__ C,
                       int M, int K, int N, int relu) {
    int idx = blockIdx.x * blockDim.x + threadIdx.x;
    if (idx >= M * N) return;
    int m = idx / N, n = idx % N;
    float s = bias ? bias[n] : 0.0f;
    const float* a = A + (size_t)m * K;
    for (int k = 0; k < K; ++k) s += a[k] * B[(size_t)k * N + n];
    if (relu) s = fmaxf(s, 0.0f);
    C[idx] = s;
}

// out[n,f] = dinv[n]^2 * xw[n,f] + b[f]   (self-loop term + bias)
__global__ void k_scatter_init(const float* xw, const float* dinv, const float* b,
                               float* out, int N, int F) {
    int idx = blockIdx.x * blockDim.x + threadIdx.x;
    if (idx >= N * F) return;
    int n = idx / F, f = idx % F;
    float di = dinv[n];
    out[idx] = di * di * xw[idx] + b[f];
}

__global__ void k_scatter_edge(const float* xw, const float* dinv,
                               const int* src, const int* dst, int nE,
                               float* out, int F) {
    int idx = blockIdx.x * blockDim.x + threadIdx.x;
    if (idx >= nE * F) return;
    int e = idx / F, f = idx % F;
    int s = src[e], d = dst[e];
    atomicAdd(&out[(size_t)d * F + f], dinv[s] * dinv[d] * xw[(size_t)s * F + f]);
}

__global__ void k_relu(float* p, int n) {
    int i = blockIdx.x * blockDim.x + threadIdx.x;
    if (i < n) p[i] = fmaxf(p[i], 0.0f);
}

// global max-pool over the 40 nodes of each graph; hcat = [h3|h5|h6]
__global__ void k_pool(const float* h3, const float* h5, const float* h6,
                       float* pooled) {
    int idx = blockIdx.x * blockDim.x + threadIdx.x;
    if (idx >= BATCH * 546) return;
    int b = idx / 546, f = idx % 546;
    const float* src; int F, fo;
    if (f < 312)      { src = h3; F = 312; fo = f; }
    else if (f < 468) { src = h5; F = 156; fo = f - 312; }
    else              { src = h6; F = 78;  fo = f - 468; }
    float m = -1.0e30f;
    for (int i = 0; i < NPG; ++i)
        m = fmaxf(m, src[(size_t)(b * NPG + i) * F + fo]);
    pooled[idx] = m;
}

// embproj[v,n] = sum_k emb[v,k]*Wih[n,k] + bih[n] + bhh[n]   (26 x 512)
__global__ void k_embproj(const float* emb, const float* Wih,
                          const float* bih, const float* bhh, float* out) {
    int idx = blockIdx.x * blockDim.x + threadIdx.x;
    if (idx >= 26 * GDIM) return;
    int v = idx / GDIM, n = idx % GDIM;
    float s = bih[n] + bhh[n];
    const float* e = emb + v * HID;
    const float* w = Wih + (size_t)n * HID;
    for (int k = 0; k < HID; ++k) s += e[k] * w[k];
    out[idx] = s;
}

// xc = [ xg | xt_f + xt_b + fc_xt_b ]
__global__ void k_xc(const float* xg, const float* xtf, const float* xtb,
                     const float* fcxt_b, float* xc) {
    int idx = blockIdx.x * blockDim.x + threadIdx.x;
    if (idx >= BATCH * 256) return;
    int b = idx / 256, j = idx % 256;
    xc[idx] = (j < 128) ? xg[b * 128 + j]
                        : (xtf[b * 128 + (j - 128)] + xtb[b * 128 + (j - 128)]
                           + fcxt_b[j - 128]);
}

// ------------------------- WMMA fragment loaders ---------------------------
// A 16x32 f16 (M x K), wave32 (ISA 7.12.2): lane l -> row l&15;
//   lanes<16: VGPR0-3 = K[kb+0..7], VGPR4-7 = K[kb+16..23]
//   lanes>=16: VGPR0-3 = K[kb+8..15], VGPR4-7 = K[kb+24..31]
__device__ inline v16h load_a16(const _Float16* H, int lane, int kb) {
    int m = lane & 15;
    int off = kb + ((lane & 16) ? 8 : 0);
    v16h a;
    ((uint4*)&a)[0] = *(const uint4*)(H + m * HID + off);
    ((uint4*)&a)[1] = *(const uint4*)(H + m * HID + off + 16);
    return a;
}

// B 32x16 f16 (K x N) from LDS stored [n][k] rowlen=128: lane l -> col nbase+(l&15);
//   lanes<16 hold K[kb+0..7],K[kb+16..23], lanes>=16 hold K[kb+8..15],K[kb+24..31]
__device__ inline v16h load_b16(const _Float16* W, int lane, int nbase, int kb) {
    int n = nbase + (lane & 15);
    int off = kb + ((lane & 16) ? 8 : 0);
    v16h b;
    ((uint4*)&b)[0] = *(const uint4*)(W + n * HID + off);
    ((uint4*)&b)[1] = *(const uint4*)(W + n * HID + off + 16);
    return b;
}

__device__ inline float sigm(float x) { return 1.0f / (1.0f + __expf(-x)); }

// --------------------------- fused LSTM kernel -----------------------------
// grid (8, 2): blockIdx.y = direction, blockIdx.x = batch chunk of 16 rows.
// 256 threads = 8 waves. Per step: gate GEMM [16,128]x[128,512] (128 WMMA) +
// fused fc_xt accumulation [16,128]x[128,128] (32 WMMA).
__global__ void k_lstm(const float* __restrict__ hidden,
                       const float* __restrict__ cell,
                       const float* __restrict__ Whh_f,
                       const float* __restrict__ Whh_b,
                       const float* __restrict__ fc_xt_w,
                       const int*   __restrict__ target,
                       const float* __restrict__ embproj,
                       float*       __restrict__ xt_ws) {
    extern __shared__ char smem[];
    _Float16* Wlds  = (_Float16*)(smem);             // [512][128] f16  131072 B
    _Float16* FxtT  = (_Float16*)(smem + 131072);    // [128][128] f16   32768 B
    _Float16* Hlds  = (_Float16*)(smem + 163840);    // [16][128]  f16    4096 B
    float*    Clds  = (float*)   (smem + 167936);    // [16][128]  f32    8192 B
    float*    Gates = (float*)   (smem + 176128);    // [16][512]  f32   32768 B
                                                     // total           208896 B
    const int tid  = threadIdx.x;
    const int lane = tid & 31;
    const int wv   = tid >> 5;
    const int dir  = blockIdx.y;
    const int m0   = blockIdx.x * 16;
    const int rowoff = (lane & 16) ? 8 : 0;

    const float* Whh = dir ? Whh_b : Whh_f;
    const float* ep0 = embproj + dir * 26 * GDIM;

    // stage Whh (f16) and h0/c0
    for (int i = tid; i < GDIM * HID; i += 256) Wlds[i] = (_Float16)Whh[i];
    for (int i = tid; i < 16 * HID; i += 256) {
        int m = i >> 7, h = i & 127;
        int g = dir * BATCH * HID + (m0 + m) * HID + h;
        Hlds[i] = (_Float16)hidden[g];
        Clds[i] = cell[g];
    }
    // prefetch first fc_xt_w block (contiguous 64KB region)
    {
        const int t0 = dir ? (TSEQ - 1) : 0;
        const char* pbase = (const char*)(fc_xt_w + ((long)t0 * 256 + dir * 128) * HID);
        __builtin_prefetch(pbase + tid * 256, 0, 3);
    }
    __syncthreads();

    v8f xacc = {};  // persistent fc_xt accumulator: M-tile 0, N-tile = wv

    for (int t = 0; t < TSEQ; ++t) {
        const int tcur = dir ? (TSEQ - 1 - t) : t;      // logical time index
        const long trow = (long)tcur * 256 + dir * 128; // fc_xt_w row base

        // stage fc_xt_w block transposed -> FxtT[n][k] (f16); prefetched a step
        // ahead so these loads hit cache instead of stalling the recurrence
        for (int i = tid; i < HID * HID; i += 256) {
            int k = i >> 7, n = i & 127;
            FxtT[n * HID + k] = (_Float16)fc_xt_w[(trow + k) * HID + n];
        }

        // gate GEMM: gates[16,512] = h @ Whh^T ; wave wv owns N-tiles wv*4..wv*4+3
        v8f g0 = {}, g1 = {}, g2 = {}, g3 = {};
        for (int kb = 0; kb < 4; ++kb) {
            v16h af = load_a16(Hlds, lane, kb * 32);
            v16h b0 = load_b16(Wlds, lane, (wv * 4 + 0) * 16, kb * 32);
            g0 = __builtin_amdgcn_wmma_f32_16x16x32_f16(false, af, false, b0, (short)0, g0, false, false);
            v16h b1 = load_b16(Wlds, lane, (wv * 4 + 1) * 16, kb * 32);
            g1 = __builtin_amdgcn_wmma_f32_16x16x32_f16(false, af, false, b1, (short)0, g1, false, false);
            v16h b2 = load_b16(Wlds, lane, (wv * 4 + 2) * 16, kb * 32);
            g2 = __builtin_amdgcn_wmma_f32_16x16x32_f16(false, af, false, b2, (short)0, g2, false, false);
            v16h b3 = load_b16(Wlds, lane, (wv * 4 + 3) * 16, kb * 32);
            g3 = __builtin_amdgcn_wmma_f32_16x16x32_f16(false, af, false, b3, (short)0, g3, false, false);
        }
        // C/D layout: VGPR r -> row r+rowoff, col = ntile*16 + (lane&15)
        {
            int c = lane & 15;
            for (int r = 0; r < 8; ++r) {
                Gates[(r + rowoff) * GDIM + (wv * 4 + 0) * 16 + c] = g0[r];
                Gates[(r + rowoff) * GDIM + (wv * 4 + 1) * 16 + c] = g1[r];
                Gates[(r + rowoff) * GDIM + (wv * 4 + 2) * 16 + c] = g2[r];
                Gates[(r + rowoff) * GDIM + (wv * 4 + 3) * 16 + c] = g3[r];
            }
        }
        __syncthreads();

        // elementwise LSTM cell update (torch gate order i,f,g,o)
        for (int i = tid; i < 16 * HID; i += 256) {
            int m = i >> 7, hc = i & 127;
            int b = m0 + m;
            int tok = target[b * TSEQ + tcur];
            const float* ep = ep0 + tok * GDIM;
            float ig = Gates[m * GDIM + hc]           + ep[hc];
            float fg = Gates[m * GDIM + 128 + hc]     + ep[128 + hc];
            float gg = Gates[m * GDIM + 256 + hc]     + ep[256 + hc];
            float og = Gates[m * GDIM + 384 + hc]     + ep[384 + hc];
            float c  = sigm(fg) * Clds[i] + sigm(ig) * tanhf(gg);
            Clds[i]  = c;
            Hlds[i]  = (_Float16)(sigm(og) * tanhf(c));
        }
        __syncthreads();

        // prefetch next step's fc_xt_w block; the WMMA burst below hides it
        if (t + 1 < TSEQ) {
            const int tnext = dir ? (TSEQ - 2 - t) : (t + 1);
            const char* pbase =
                (const char*)(fc_xt_w + ((long)tnext * 256 + dir * 128) * HID);
            __builtin_prefetch(pbase + tid * 256, 0, 3);
        }

        // fused fc_xt: xacc += h_t @ fc_xt_w[trow:trow+128, :]   (wave wv -> N-tile wv)
        for (int kb = 0; kb < 4; ++kb) {
            v16h af = load_a16(Hlds, lane, kb * 32);
            v16h bf = load_b16(FxtT, lane, wv * 16, kb * 32);
            xacc = __builtin_amdgcn_wmma_f32_16x16x32_f16(false, af, false, bf, (short)0, xacc, false, false);
        }
        __syncthreads();   // protect Hlds / FxtT before next-iter restage
    }

    // write xt partials: xt_ws[dir][b][j]
    {
        int c = lane & 15;
        for (int r = 0; r < 8; ++r)
            xt_ws[dir * BATCH * HID + (m0 + r + rowoff) * HID + wv * 16 + c] = xacc[r];
    }
}

// ------------------------------ host side ----------------------------------

static inline int cdiv(long a, int b) { return (int)((a + b - 1) / b); }

extern "C" void kernel_launch(void* const* d_in, const int* in_sizes, int n_in,
                              void* d_out, int out_size, void* d_ws, size_t ws_size,
                              hipStream_t stream) {
    // input order: x, hidden, cell, <27 params in dict order>, e1, e2, e3, batch, target
    const float* x       = (const float*)d_in[0];
    const float* hidden  = (const float*)d_in[1];
    const float* cell    = (const float*)d_in[2];
    const float* W1      = (const float*)d_in[3];
    const float* b1      = (const float*)d_in[4];
    const float* W2      = (const float*)d_in[5];
    const float* b2      = (const float*)d_in[6];
    const float* W3      = (const float*)d_in[7];
    const float* b3      = (const float*)d_in[8];
    const float* fc_g1_w = (const float*)d_in[9];
    const float* fc_g1_b = (const float*)d_in[10];
    const float* fc_g2_w = (const float*)d_in[11];
    const float* fc_g2_b = (const float*)d_in[12];
    const float* emb     = (const float*)d_in[13];
    const float* Wih_f   = (const float*)d_in[14];
    const float* Whh_f   = (const float*)d_in[15];
    const float* bih_f   = (const float*)d_in[16];
    const float* bhh_f   = (const float*)d_in[17];
    const float* Wih_b   = (const float*)d_in[18];
    const float* Whh_b   = (const float*)d_in[19];
    const float* bih_b   = (const float*)d_in[20];
    const float* bhh_b   = (const float*)d_in[21];
    const float* fc_xt_w = (const float*)d_in[22];
    const float* fc_xt_b = (const float*)d_in[23];
    const float* fc1_w   = (const float*)d_in[24];
    const float* fc1_b   = (const float*)d_in[25];
    const float* fc2_w   = (const float*)d_in[26];
    const float* fc2_b   = (const float*)d_in[27];
    const float* out_w   = (const float*)d_in[28];
    const float* out_b   = (const float*)d_in[29];
    const int*   e1      = (const int*)d_in[30];
    const int*   e2      = (const int*)d_in[31];
    const int*   e3      = (const int*)d_in[32];
    const int*   target  = (const int*)d_in[34];
    const int E1 = in_sizes[30] / 2, E2 = in_sizes[31] / 2, E3 = in_sizes[32] / 2;

    // workspace layout (f32, 256B-aligned bump allocator)
    char* wsb = (char*)d_ws;
    size_t off = 0;
    auto alloc = [&](size_t nfloats) -> float* {
        float* p = (float*)(wsb + off);
        off += (nfloats * sizeof(float) + 255) & ~(size_t)255;
        return p;
    };
    float* xw1    = alloc((size_t)N_NODES * 78);
    float* xwbuf  = alloc((size_t)N_NODES * 312);
    float* h1     = alloc((size_t)N_NODES * 78);
    float* h2     = alloc((size_t)N_NODES * 156);
    float* h3     = alloc((size_t)N_NODES * 312);
    float* h4     = alloc((size_t)N_NODES * 78);
    float* h5     = alloc((size_t)N_NODES * 156);
    float* h6     = alloc((size_t)N_NODES * 78);
    float* dinv1  = alloc(N_NODES);
    float* dinv2  = alloc(N_NODES);
    float* dinv3  = alloc(N_NODES);
    float* pooled = alloc((size_t)BATCH * 546);
    float* xg1    = alloc((size_t)BATCH * 1024);
    float* xg     = alloc((size_t)BATCH * 128);
    float* embprj = alloc((size_t)2 * 26 * GDIM);
    float* xt_ws  = alloc((size_t)2 * BATCH * HID);
    float* xc     = alloc((size_t)BATCH * 256);
    float* t1     = alloc((size_t)BATCH * 1024);
    float* t2     = alloc((size_t)BATCH * 512);
    (void)ws_size; (void)n_in; (void)out_size;

    const int TB = 256;
    // degrees -> dinv for the three edge sets (deg = in-degree + self loop)
    k_fill<<<cdiv(N_NODES, TB), TB, 0, stream>>>(dinv1, 1.0f, N_NODES);
    k_fill<<<cdiv(N_NODES, TB), TB, 0, stream>>>(dinv2, 1.0f, N_NODES);
    k_fill<<<cdiv(N_NODES, TB), TB, 0, stream>>>(dinv3, 1.0f, N_NODES);
    k_deg<<<cdiv(E1, TB), TB, 0, stream>>>(e1 + E1, E1, dinv1);
    k_deg<<<cdiv(E2, TB), TB, 0, stream>>>(e2 + E2, E2, dinv2);
    k_deg<<<cdiv(E3, TB), TB, 0, stream>>>(e3 + E3, E3, dinv3);
    k_rsqrt<<<cdiv(N_NODES, TB), TB, 0, stream>>>(dinv1, N_NODES);
    k_rsqrt<<<cdiv(N_NODES, TB), TB, 0, stream>>>(dinv2, N_NODES);
    k_rsqrt<<<cdiv(N_NODES, TB), TB, 0, stream>>>(dinv3, N_NODES);

    // GCN layer helper (xw precomputed): scatter_init + edge scatter + relu
    auto gcn = [&](const float* xw, const float* dinv, const int* e, int nE,
                   const float* bias, float* out, int F) {
        k_scatter_init<<<cdiv((long)N_NODES * F, TB), TB, 0, stream>>>(xw, dinv, bias, out, N_NODES, F);
        k_scatter_edge<<<cdiv((long)nE * F, TB), TB, 0, stream>>>(xw, dinv, e, e + nE, nE, out, F);
        k_relu<<<cdiv((long)N_NODES * F, TB), TB, 0, stream>>>(out, N_NODES * F);
    };

    // xw1 = x @ W1 (reused for h1, h4, h6)
    k_gemm<<<cdiv((long)N_NODES * 78, TB), TB, 0, stream>>>(x, W1, nullptr, xw1, N_NODES, 78, 78, 0);
    gcn(xw1, dinv1, e1, E1, b1, h1, 78);
    k_gemm<<<cdiv((long)N_NODES * 156, TB), TB, 0, stream>>>(h1, W2, nullptr, xwbuf, N_NODES, 78, 156, 0);
    gcn(xwbuf, dinv1, e1, E1, b2, h2, 156);
    k_gemm<<<cdiv((long)N_NODES * 312, TB), TB, 0, stream>>>(h2, W3, nullptr, xwbuf, N_NODES, 156, 312, 0);
    gcn(xwbuf, dinv1, e1, E1, b3, h3, 312);
    gcn(xw1, dinv2, e2, E2, b1, h4, 78);
    k_gemm<<<cdiv((long)N_NODES * 156, TB), TB, 0, stream>>>(h4, W2, nullptr, xwbuf, N_NODES, 78, 156, 0);
    gcn(xwbuf, dinv2, e2, E2, b2, h5, 156);
    gcn(xw1, dinv3, e3, E3, b1, h6, 78);

    // pool + graph FCs
    k_pool<<<cdiv(BATCH * 546, TB), TB, 0, stream>>>(h3, h5, h6, pooled);
    k_gemm<<<cdiv(BATCH * 1024, TB), TB, 0, stream>>>(pooled, fc_g1_w, fc_g1_b, xg1, BATCH, 546, 1024, 1);
    k_gemm<<<cdiv(BATCH * 128, TB), TB, 0, stream>>>(xg1, fc_g2_w, fc_g2_b, xg, BATCH, 1024, 128, 0);

    // embedding projection tables (fold bih+bhh)
    k_embproj<<<cdiv(26 * GDIM, TB), TB, 0, stream>>>(emb, Wih_f, bih_f, bhh_f, embprj);
    k_embproj<<<cdiv(26 * GDIM, TB), TB, 0, stream>>>(emb, Wih_b, bih_b, bhh_b, embprj + 26 * GDIM);

    // fused bidirectional LSTM + fc_xt (WMMA): grid (8 batch chunks, 2 dirs)
    const size_t lstm_lds = 131072 + 32768 + 4096 + 8192 + 32768; // 208896 B
    k_lstm<<<dim3(8, 2), 256, lstm_lds, stream>>>(hidden, cell, Whh_f, Whh_b,
                                                  fc_xt_w, target, embprj, xt_ws);

    // head
    k_xc<<<cdiv(BATCH * 256, TB), TB, 0, stream>>>(xg, xt_ws, xt_ws + BATCH * HID, fc_xt_b, xc);
    k_gemm<<<cdiv(BATCH * 1024, TB), TB, 0, stream>>>(xc, fc1_w, fc1_b, t1, BATCH, 256, 1024, 1);
    k_gemm<<<cdiv(BATCH * 512, TB), TB, 0, stream>>>(t1, fc2_w, fc2_b, t2, BATCH, 1024, 512, 1);
    k_gemm<<<cdiv(BATCH, TB), TB, 0, stream>>>(t2, out_w, out_b, (float*)d_out, BATCH, 512, 1, 0);
}
